// BertSelfAttention_35433480192573
// MI455X (gfx1250) — matrix operand order repaired and checked by reading, not compiled, not run on hardware
//
#include <hip/hip_runtime.h>
#include <hip/hip_bf16.h>

// ---------------------------------------------------------------------------
// BERT self-attention for MI455X (gfx1250, wave32, WMMA bf16 16x16x32)
// B=16, S=512, H=1024, nh=16, dh=64
// out = concat(ctx [B,S,H] f32, scores [B,nh,S,S] f32)
// workspace: Q bf16 (16MB) + K bf16 (16MB) + Vt bf16 (16MB) = 48MB
// Fused attention kernel: scores written to HBM once, never re-read.
// ---------------------------------------------------------------------------

#define BB   16
#define SS   512
#define HH   1024
#define NH   16
#define DH   64
#define MM   (BB * SS)   // 8192

typedef __attribute__((ext_vector_type(16))) __bf16 v16bf;
typedef __attribute__((ext_vector_type(8)))  float  v8f;
typedef __attribute__((ext_vector_type(4)))  unsigned int v4u;

union Frag {
    v4u   q[2];
    v16bf v;
};

__device__ __forceinline__ unsigned short f32_to_bf16(float f) {
    union { float f; unsigned int u; } c;
    c.f = f;
    unsigned int u = c.u;
    u = (u + 0x7FFFu + ((u >> 16) & 1u)) >> 16;   // round to nearest even
    return (unsigned short)u;
}

// ---------------------------------------------------------------------------
// Kernel 1: out[m,n] = X[m,:] . W[n,:] + bias[n]   (x @ W.T + b)
// M=8192, N=1024, K=1024.  Block tile 128x64, 256 thr = 8 waves (4x2),
// each wave 32x32 = 2x2 WMMA 16x16 tiles, K stepped by 32.
// TRANSPOSE=0: bf16 out [b,h,s,d]   TRANSPOSE=1: bf16 out [b,h,d,s] (for V)
// ---------------------------------------------------------------------------
template <int TRANSPOSE>
__global__ __launch_bounds__(256)
void qkv_gemm(const float* __restrict__ X, const float* __restrict__ W,
              const float* __restrict__ bias, unsigned short* __restrict__ out) {
    __shared__ __align__(16) unsigned short ldsA[128 * 40]; // 128 rows x 32k, pad 8
    __shared__ __align__(16) unsigned short ldsB[64 * 40];  //  64 rows x 32k, pad 8

    const int t    = threadIdx.x;
    const int lane = t & 31;
    const int wave = t >> 5;
    const int hi   = lane >> 4;
    const int ln   = lane & 15;
    const int wm   = wave & 3;
    const int wn   = wave >> 2;
    const int m0   = blockIdx.x * 128;
    const int n0   = blockIdx.y * 64;

    v8f acc[2][2] = {};

    for (int k0 = 0; k0 < HH; k0 += 32) {
        __syncthreads();
        #pragma unroll
        for (int i = 0; i < 4; ++i) {              // A tile 128x32
            int e   = t + i * 256;
            int row = e >> 3;
            int kk  = (e & 7) << 2;
            const float* src = X + (size_t)(m0 + row) * HH + k0 + kk;
            const float4 f = *(const float4*)src;
            if (k0 + 32 < HH) __builtin_prefetch(src + 32, 0, 1);
            unsigned short* p = &ldsA[row * 40 + kk];
            p[0] = f32_to_bf16(f.x); p[1] = f32_to_bf16(f.y);
            p[2] = f32_to_bf16(f.z); p[3] = f32_to_bf16(f.w);
        }
        #pragma unroll
        for (int i = 0; i < 2; ++i) {              // B tile 64x32 of W[n,k]
            int e   = t + i * 256;
            int row = e >> 3;
            int kk  = (e & 7) << 2;
            const float* src = W + (size_t)(n0 + row) * HH + k0 + kk;
            const float4 f = *(const float4*)src;
            if (k0 + 32 < HH) __builtin_prefetch(src + 32, 0, 1);
            unsigned short* p = &ldsB[row * 40 + kk];
            p[0] = f32_to_bf16(f.x); p[1] = f32_to_bf16(f.y);
            p[2] = f32_to_bf16(f.z); p[3] = f32_to_bf16(f.w);
        }
        __syncthreads();

        Frag a[2], b[2];
        #pragma unroll
        for (int mt = 0; mt < 2; ++mt) {
            const unsigned short* pr = &ldsA[(wm * 32 + mt * 16 + ln) * 40 + hi * 8];
            a[mt].q[0] = *(const v4u*)(pr);
            a[mt].q[1] = *(const v4u*)(pr + 16);
        }
        #pragma unroll
        for (int nt = 0; nt < 2; ++nt) {
            const unsigned short* pr = &ldsB[(wn * 32 + nt * 16 + ln) * 40 + hi * 8];
            b[nt].q[0] = *(const v4u*)(pr);
            b[nt].q[1] = *(const v4u*)(pr + 16);
        }
        #pragma unroll
        for (int mt = 0; mt < 2; ++mt)
            #pragma unroll
            for (int nt = 0; nt < 2; ++nt)
                acc[mt][nt] = __builtin_amdgcn_wmma_f32_16x16x32_bf16(
                    false, a[mt].v, false, b[nt].v, (short)0, acc[mt][nt],
                    false, false);
    }

    #pragma unroll
    for (int nt = 0; nt < 2; ++nt) {
        int n = n0 + wn * 32 + nt * 16 + ln;
        float bia = bias[n];
        int h = n >> 6, d = n & 63;
        #pragma unroll
        for (int mt = 0; mt < 2; ++mt) {
            #pragma unroll
            for (int j = 0; j < 8; ++j) {
                int m  = m0 + wm * 32 + mt * 16 + j + hi * 8;
                int bb = m >> 9, s = m & 511;
                float val = acc[mt][nt][j] + bia;
                size_t addr;
                if (TRANSPOSE)
                    addr = ((size_t)((bb * NH + h) * DH + d)) * SS + s;
                else
                    addr = ((size_t)((bb * NH + h) * SS + s)) * DH + d;
                out[addr] = f32_to_bf16(val);
            }
        }
    }
}

// ---------------------------------------------------------------------------
// Kernel 2 (fused attention): per (b,h, 16-query strip):
//   Phase 1: scores strip 16x512 = Q@K^T * 0.125, mask|diag -> -inf.
//            Written to global (output) AND to LDS (never re-read from HBM).
//            8 waves, each 16q x 64k (4 WMMA n-tiles, 2 k-steps).
//   Phase 2: row softmax from LDS; p = exp*inv_sum*group_prob -> bf16 LDS
//            tile in WMMA A-layout (padded stride 520).
//   Phase 3: ctx[16x64] = P @ V (transposed V [b,h,d,s]); 8 waves =
//            4 d-tiles x 2 K-halves; cross-wave reduction via LDS.
// ---------------------------------------------------------------------------
__global__ __launch_bounds__(256)
void attn_fused(const unsigned short* __restrict__ qb,
                const unsigned short* __restrict__ kb,
                const int* __restrict__ am, const float* __restrict__ gp,
                const unsigned short* __restrict__ vtb,
                float* __restrict__ sc, float* __restrict__ ctx) {
    __shared__ __align__(16) float strip[16 * 520];          // 33.3 KB (reused)
    __shared__ __align__(16) unsigned short probs[16 * 520]; // 16.6 KB
    __shared__ float red[16][16];
    __shared__ float rowMax[16];
    __shared__ float rowInv[16];

    const int t    = threadIdx.x;
    const int lane = t & 31;
    const int wave = t >> 5;
    const int hi   = lane >> 4;
    const int ln   = lane & 15;
    const int qt   = blockIdx.x;      // 0..31
    const int bh   = blockIdx.y;      // 0..255
    const int b    = bh >> 4, h = bh & 15;
    const int q0   = qt * 16;

    const unsigned short* qp = qb + (size_t)bh * SS * DH;
    const unsigned short* kp = kb + (size_t)bh * SS * DH;

    // ---- Phase 1: score strip 16 x 512 ----
    {
        const int kb0  = wave * 64;           // wave's 64 key columns
        const int qrow = q0 + ln;
        v8f acc[4] = {};
        #pragma unroll
        for (int k0 = 0; k0 < DH; k0 += 32) {
            Frag a;
            const unsigned short* pa = qp + (size_t)qrow * DH + k0 + hi * 8;
            a.q[0] = *(const v4u*)pa;
            a.q[1] = *(const v4u*)(pa + 16);
            #pragma unroll
            for (int nt = 0; nt < 4; ++nt) {
                int krow = kb0 + nt * 16 + ln;
                const unsigned short* pb = kp + (size_t)krow * DH + k0 + hi * 8;
                Frag bf;
                bf.q[0] = *(const v4u*)pb;
                bf.q[1] = *(const v4u*)(pb + 16);
                acc[nt] = __builtin_amdgcn_wmma_f32_16x16x32_bf16(
                    false, a.v, false, bf.v, (short)0, acc[nt], false, false);
            }
        }
        #pragma unroll
        for (int nt = 0; nt < 4; ++nt) {
            int kcol = kb0 + nt * 16 + ln;
            #pragma unroll
            for (int j = 0; j < 8; ++j) {
                int qr = j + hi * 8;          // local row 0..15
                int q  = q0 + qr;
                int keep = am[((size_t)b * SS + q) * SS + kcol] | (int)(q == kcol);
                float val = acc[nt][j] * 0.125f;
                val = keep ? val : -__builtin_inff();
                sc[((size_t)bh * SS + q) * SS + kcol] = val;
                strip[qr * 520 + kcol] = val;
            }
        }
    }
    __syncthreads();

    // ---- Phase 2: softmax (16 threads per row, 32 cols each) ----
    {
        const int r   = t >> 4;               // 0..15
        const int seg = t & 15;               // col = seg + 16*c -> coalesced
        float lmax = -__builtin_inff();
        for (int c = 0; c < 32; ++c)
            lmax = fmaxf(lmax, strip[r * 520 + seg + (c << 4)]);
        red[r][seg] = lmax;
        __syncthreads();
        if (seg == 0) {
            float m = red[r][0];
            #pragma unroll
            for (int i = 1; i < 16; ++i) m = fmaxf(m, red[r][i]);
            rowMax[r] = m;
        }
        __syncthreads();
        const float m = rowMax[r];
        float lsum = 0.0f;
        for (int c = 0; c < 32; ++c)
            lsum += __expf(strip[r * 520 + seg + (c << 4)] - m);
        red[r][seg] = lsum;
        __syncthreads();
        if (seg == 0) {
            float s = red[r][0];
            #pragma unroll
            for (int i = 1; i < 16; ++i) s += red[r][i];
            rowInv[r] = 1.0f / s;
        }
        __syncthreads();
        const float inv = rowInv[r];
        const float* gprow = gp + ((size_t)b * SS + q0 + r) * SS;
        for (int c = 0; c < 32; ++c) {
            int col = seg + (c << 4);
            float p = __expf(strip[r * 520 + col] - m) * inv * gprow[col];
            probs[r * 520 + col] = f32_to_bf16(p);
        }
    }
    __syncthreads();   // strip is dead from here; reused as reduction buffer

    // ---- Phase 3: ctx[16 x 64] = P[16x512] @ V[512x64] ----
    const int ni = wave & 3;                  // d tile of 16
    const int kh = wave >> 2;                 // K half: [kh*256, kh*256+256)
    const unsigned short* vtp = vtb + (size_t)bh * DH * SS;
    const int d = ni * 16 + ln;

    v8f acc = {};
    #pragma unroll 4
    for (int kk = 0; kk < 256; kk += 32) {
        int k0 = kh * 256 + kk;
        Frag a, bf;
        const unsigned short* pa = &probs[ln * 520 + k0 + hi * 8];
        a.q[0] = *(const v4u*)pa;
        a.q[1] = *(const v4u*)(pa + 16);
        const unsigned short* pb = vtp + (size_t)d * SS + k0 + hi * 8;
        bf.q[0] = *(const v4u*)pb;
        bf.q[1] = *(const v4u*)(pb + 16);
        acc = __builtin_amdgcn_wmma_f32_16x16x32_bf16(
            false, a.v, false, bf.v, (short)0, acc, false, false);
    }

    // cross-wave reduction of the two K-halves through (reused) strip LDS
    float* part = strip;                      // 4 tiles x 16 x 16 = 1024 floats
    if (kh == 1) {
        #pragma unroll
        for (int j = 0; j < 8; ++j)
            part[ni * 256 + (j + hi * 8) * 16 + ln] = acc[j];
    }
    __syncthreads();
    if (kh == 0) {
        #pragma unroll
        for (int j = 0; j < 8; ++j) {
            float v = acc[j] + part[ni * 256 + (j + hi * 8) * 16 + ln];
            int q = q0 + j + hi * 8;
            ctx[((size_t)b * SS + q) * HH + h * DH + d] = v;
        }
    }
}

// ---------------------------------------------------------------------------
extern "C" void kernel_launch(void* const* d_in, const int* in_sizes, int n_in,
                              void* d_out, int out_size, void* d_ws, size_t ws_size,
                              hipStream_t stream) {
    const float* hs = (const float*)d_in[0];
    const int*   am = (const int*)  d_in[1];
    const float* gp = (const float*)d_in[2];
    const float* Wq = (const float*)d_in[3];
    const float* bq = (const float*)d_in[4];
    const float* Wk = (const float*)d_in[5];
    const float* bk = (const float*)d_in[6];
    const float* Wv = (const float*)d_in[7];
    const float* bv = (const float*)d_in[8];

    float* ctx = (float*)d_out;                       // [B,S,H]
    float* sc  = ctx + (size_t)BB * SS * HH;          // [B,nh,S,S]

    unsigned short* qb  = (unsigned short*)d_ws;      // Q bf16 [b,h,s,d]
    unsigned short* kb  = qb + (size_t)MM * HH;       // K bf16 [b,h,s,d]
    unsigned short* vtb = kb + (size_t)MM * HH;       // V bf16 [b,h,d,s]

    dim3 gQKV(MM / 128, HH / 64);                     // 64 x 16
    qkv_gemm<0><<<gQKV, 256, 0, stream>>>(hs, Wq, bq, qb);
    qkv_gemm<0><<<gQKV, 256, 0, stream>>>(hs, Wk, bk, kb);
    qkv_gemm<1><<<gQKV, 256, 0, stream>>>(hs, Wv, bv, vtb);

    dim3 gAtt(SS / 16, BB * NH);                      // 32 x 256
    attn_fused<<<gAtt, 256, 0, stream>>>(qb, kb, am, gp, vtb, sc, ctx);
}
